// Pseudo_MHSA_Block_21328807592207
// MI455X (gfx1250) — matrix operand order repaired and checked by reading, not compiled, not run on hardware
//
#include <hip/hip_runtime.h>
#include <stdint.h>

// ---------------------------------------------------------------------------
// Transformer block (pseudo-MHSA + MLP) for MI455X (gfx1250, wave32, WMMA).
// All matmuls on v_wmma_f32_16x16x32_bf16 (f32 accumulate). GEMM tiles are
// staged into LDS with double-buffered GLOBAL_LOAD_ASYNC_TO_LDS_B128
// (s_wait_asynccnt-synchronized), falling back to synchronous b128 staging if
// the async builtins are absent. Weights are pre-transposed at bf16-conversion
// time so B staging is a contiguous copy. Attention is flash-style (online
// softmax), one wave per 16-query tile, V pre-transposed to [B,H,D,S] so every
// WMMA operand is loaded as two contiguous 16B b128s.
// ---------------------------------------------------------------------------

#define B_  4
#define S_  2048
#define E_  768
#define H_  12
#define D_  64
#define F_  3072
#define BS_ (B_*S_)

typedef unsigned short u16;
typedef __attribute__((ext_vector_type(16))) __bf16 bf16x16;
typedef __attribute__((ext_vector_type(8)))  float  f32x8;
typedef __attribute__((ext_vector_type(4)))  int    i32x4;

union FragB { uint4 q[2]; bf16x16 v; };

__device__ __forceinline__ u16 f2bf(float f) {
  uint32_t u = __float_as_uint(f);
  u += 0x7FFFu + ((u >> 16) & 1u);          // round-to-nearest-even
  return (u16)(u >> 16);
}

#if __has_builtin(__builtin_amdgcn_global_load_async_to_lds_b128) && \
    __has_builtin(__builtin_amdgcn_s_wait_asynccnt)
#define USE_ASYNC_LDS 1
#else
#define USE_ASYNC_LDS 0
#endif

#if USE_ASYNC_LDS
// 16-byte global -> LDS async copy (per-lane addresses, ASYNCcnt-tracked).
// Builtin takes int4-vector pointers: AS1 source, AS3 destination. The
// generic->LDS cast goes through uintptr_t: on gfx1250 the low 32 bits of a
// flat LDS-aperture address are the LDS byte offset (ISA 10.2 aperture rules).
__device__ __forceinline__ void async_cp16(const u16* g, const u16* l) {
  __builtin_amdgcn_global_load_async_to_lds_b128(
      (__attribute__((address_space(1))) i32x4*)(uintptr_t)g,
      (__attribute__((address_space(3))) i32x4*)(uint32_t)(uintptr_t)l,
      0, 0);
}
#endif

// ---------------------------------------------------------------------------
// f32 -> bf16 conversion (plain)
// ---------------------------------------------------------------------------
__global__ void cvt_bf16(const float* __restrict__ in, u16* __restrict__ out, int n) {
  int i = blockIdx.x * 256 + threadIdx.x;
  if (i < n) out[i] = f2bf(in[i]);
}

// f32[batch][K][N] -> bf16 transposed [batch][N][K] (weights)
__global__ void cvt_bf16_t(const float* __restrict__ in, u16* __restrict__ out,
                           int K, int N, int total) {
  int idx = blockIdx.x * 256 + threadIdx.x;
  if (idx >= total) return;
  const int kn = K * N;
  const int b = idx / kn, rem = idx - b * kn;
  const int k = rem / N, n = rem - k * N;
  out[(size_t)b * kn + (size_t)n * K + k] = f2bf(in[idx]);
}

// ---------------------------------------------------------------------------
// LayerNorm over E=768, one row per 256-thread block, bf16 output
// ---------------------------------------------------------------------------
__global__ __launch_bounds__(256) void ln_bf16(
    const float* __restrict__ x, const float* __restrict__ scale,
    const float* __restrict__ bias, u16* __restrict__ out)
{
  __shared__ float s1[256], s2[256];
  const int row = blockIdx.x;
  const float* xr = x + (size_t)row * E_;
  float a = 0.f, b = 0.f, vals[3];
#pragma unroll
  for (int i = 0; i < 3; ++i) {
    float v = xr[threadIdx.x + i * 256];
    vals[i] = v; a += v; b += v * v;
  }
  s1[threadIdx.x] = a; s2[threadIdx.x] = b;
  __syncthreads();
  for (int off = 128; off > 0; off >>= 1) {
    if (threadIdx.x < off) {
      s1[threadIdx.x] += s1[threadIdx.x + off];
      s2[threadIdx.x] += s2[threadIdx.x + off];
    }
    __syncthreads();
  }
  const float mu   = s1[0] * (1.0f / E_);
  const float var  = s2[0] * (1.0f / E_) - mu * mu;
  const float rstd = rsqrtf(var + 1e-5f);
  u16* orow = out + (size_t)row * E_;
#pragma unroll
  for (int i = 0; i < 3; ++i) {
    const int c = threadIdx.x + i * 256;
    orow[c] = f2bf((vals[i] - mu) * rstd * scale[c] + bias[c]);
  }
}

// ---------------------------------------------------------------------------
// bf16 WMMA GEMM: C[M,N] = A[M,K] * Bt[N,K]^T (+bias)(+GELU)(+residual)
// Block tile 128x64, 8 waves (4x2), each wave 32x32 = 2x2 WMMA tiles, K-step
// 32. A and Bt tiles staged in LDS; double-buffered async-to-LDS when
// available. blockIdx.z batches the 12 per-head DxD GEMMs.
// ---------------------------------------------------------------------------
template<bool BIAS, bool GELU, bool RES, bool OUTBF>
__global__ __launch_bounds__(256) void gemm_bf16(
    const u16* __restrict__ A, int lda, long long aBatch,
    const u16* __restrict__ Bt, int ldbt, long long bBatch,
    const float* __restrict__ bias,
    const float* __restrict__ resid, int ldres, long long rBatch,
    u16* __restrict__ Cbf, float* __restrict__ Cf, int ldc, long long cBatch,
    int M, int N, int K)
{
  __shared__ __align__(16) u16 As [2][128 * 32];
  __shared__ __align__(16) u16 BsT[2][64 * 32];

  const int tid  = threadIdx.x;
  const int lane = tid & 31;
  const int wave = tid >> 5;
  const int wm   = wave >> 1;          // 0..3 : M offset *32
  const int wn   = wave & 1;           // 0..1 : N offset *32
  const int hi   = (lane >> 4) & 1;
  const int l15  = lane & 15;

  const long long zb = blockIdx.z;
  A  += zb * aBatch;
  Bt += zb * bBatch;
  const float* res = RES ? (resid + zb * rBatch) : nullptr;
  u16*   cb = Cbf + (OUTBF ? zb * cBatch : 0);
  float* cf = Cf  + (OUTBF ? 0 : zb * cBatch);

  const int bm = blockIdx.y * 128;
  const int bn = blockIdx.x * 64;

  f32x8 acc[2][2] = {};

  // staging assignments (256 threads)
  const int arow = tid >> 1;           // 0..127
  const int acol = (tid & 1) * 16;     // 0 / 16 (elems); chunks at +0, +8
  const int brow = tid >> 2;           // 0..63  (n)
  const int bseg = (tid & 3) * 8;      // 0,8,16,24 (k elems; 16B chunk)

  const int koffA = hi ? 8 : 0;
  const int koffB = hi ? 8 : 0;

  auto compute = [&](const u16* as, const u16* bst) {
    FragB af[2], bfm[2];
#pragma unroll
    for (int mt = 0; mt < 2; ++mt) {
      const int r = wm * 32 + mt * 16 + l15;
      af[mt].q[0] = *(const uint4*)&as[r * 32 + koffA];
      af[mt].q[1] = *(const uint4*)&as[r * 32 + koffA + 16];
    }
#pragma unroll
    for (int nt = 0; nt < 2; ++nt) {
      const int n = wn * 32 + nt * 16 + l15;
      bfm[nt].q[0] = *(const uint4*)&bst[n * 32 + koffB];
      bfm[nt].q[1] = *(const uint4*)&bst[n * 32 + koffB + 16];
    }
#pragma unroll
    for (int mt = 0; mt < 2; ++mt)
#pragma unroll
      for (int nt = 0; nt < 2; ++nt)
        acc[mt][nt] = __builtin_amdgcn_wmma_f32_16x16x32_bf16(
            false, af[mt].v, false, bfm[nt].v, (short)0, acc[mt][nt], false, false);
  };

#if USE_ASYNC_LDS
  auto issue = [&](int buf, int k0) {
    const u16* ag = A  + (size_t)(bm + arow) * lda  + k0 + acol;
    const u16* bg = Bt + (size_t)(bn + brow) * ldbt + k0 + bseg;
    async_cp16(ag,     &As [buf][arow * 32 + acol]);
    async_cp16(ag + 8, &As [buf][arow * 32 + acol + 8]);
    async_cp16(bg,     &BsT[buf][brow * 32 + bseg]);
  };
  issue(0, 0);
  int cur = 0;
  for (int k0 = 0; k0 < K; k0 += 32) {
    __builtin_amdgcn_s_wait_asynccnt(0);   // this wave's copies landed in LDS
    __syncthreads();                       // all waves' copies + prior reads done
    if (k0 + 32 < K) issue(cur ^ 1, k0 + 32);
    compute(As[cur], BsT[cur]);
    cur ^= 1;
  }
#else
  for (int k0 = 0; k0 < K; k0 += 32) {
    const u16* ag = A  + (size_t)(bm + arow) * lda  + k0 + acol;
    const u16* bg = Bt + (size_t)(bn + brow) * ldbt + k0 + bseg;
    *(uint4*)&As [0][arow * 32 + acol]     = *(const uint4*)(ag);
    *(uint4*)&As [0][arow * 32 + acol + 8] = *(const uint4*)(ag + 8);
    *(uint4*)&BsT[0][brow * 32 + bseg]     = *(const uint4*)(bg);
    if (k0 + 32 < K) {
      __builtin_prefetch(ag + 32, 0, 1);
      __builtin_prefetch(bg + 32, 0, 1);
    }
    __syncthreads();
    compute(As[0], BsT[0]);
    __syncthreads();
  }
#endif

#pragma unroll
  for (int mt = 0; mt < 2; ++mt) {
#pragma unroll
    for (int nt = 0; nt < 2; ++nt) {
      const int col = bn + wn * 32 + nt * 16 + l15;
#pragma unroll
      for (int r = 0; r < 8; ++r) {
        const int row = bm + wm * 32 + mt * 16 + hi * 8 + r;
        float v = acc[mt][nt][r];
        if (BIAS) v += bias[col];
        if (GELU) v = 0.5f * v * (1.0f + erff(v * 0.70710678118654752f));
        if (RES)  v += res[(size_t)row * ldres + col];
        if (OUTBF) cb[(size_t)row * ldc + col] = f2bf(v);
        else       cf[(size_t)row * ldc + col] = v;
      }
    }
  }
}

// ---------------------------------------------------------------------------
// Transpose xv [B,S,H,D] -> xvT [B,H,D,S]
// ---------------------------------------------------------------------------
__global__ void xv_transpose(const u16* __restrict__ in, u16* __restrict__ out) {
  size_t idx = (size_t)blockIdx.x * 256 + threadIdx.x;   // over BS_*E_
  const int    col = (int)(idx % E_);
  const size_t row = idx / E_;                            // b*S + s
  const int h = col >> 6, d = col & 63;
  const size_t bb = row / S_, s = row % S_;
  out[(((bb * H_ + h) * D_) + d) * S_ + s] = in[idx];
}

// ---------------------------------------------------------------------------
// Flash attention: one wave per 16-query tile per (b,h).
// ---------------------------------------------------------------------------
__global__ __launch_bounds__(32) void attn_flash(
    const u16* __restrict__ xq, const u16* __restrict__ xk,
    const u16* __restrict__ xvT, u16* __restrict__ ao)
{
  __shared__ __align__(16) u16 P[16 * 32];
  const int lane = threadIdx.x & 31;
  const int hi   = (lane >> 4) & 1;
  const int l15  = lane & 15;
  const int qt = blockIdx.x, h = blockIdx.y, b = blockIdx.z;
  const size_t rowQ = (size_t)b * S_ + qt * 16 + l15;
  const int colH = h * D_;
  const int koffA = hi ? 8 : 0;
  const int koffB = hi ? 8 : 0;

  FragB qa[2];
#pragma unroll
  for (int kc = 0; kc < 2; ++kc) {
    const u16* p = xq + rowQ * E_ + colH + kc * 32 + koffA;
    qa[kc].q[0] = *(const uint4*)(p);
    qa[kc].q[1] = *(const uint4*)(p + 16);
  }

  float m[8], l[8];
  f32x8 accO[4] = {};
#pragma unroll
  for (int r = 0; r < 8; ++r) { m[r] = -3.0e38f; l[r] = 0.f; }

  const size_t kbase = (size_t)b * S_;
  const size_t vbase = (size_t)(b * H_ + h) * D_ * S_;

  for (int j = 0; j < S_; j += 32) {
    f32x8 s[2] = {};
#pragma unroll
    for (int nh = 0; nh < 2; ++nh) {
#pragma unroll
      for (int kc = 0; kc < 2; ++kc) {
        FragB kb;
        const u16* p = xk + (kbase + j + nh * 16 + l15) * E_ + colH + kc * 32 + koffB;
        kb.q[0] = *(const uint4*)(p);
        kb.q[1] = *(const uint4*)(p + 16);
        s[nh] = __builtin_amdgcn_wmma_f32_16x16x32_bf16(
            false, qa[kc].v, false, kb.v, (short)0, s[nh], false, false);
      }
    }
    float alpha[8];
#pragma unroll
    for (int r = 0; r < 8; ++r) {
      s[0][r] *= (1.0f / D_);
      s[1][r] *= (1.0f / D_);
      float t = fmaxf(s[0][r], s[1][r]);
#pragma unroll
      for (int off = 1; off < 16; off <<= 1)
        t = fmaxf(t, __shfl_xor(t, off, 32));
      const float mnew = fmaxf(m[r], t);
      alpha[r] = __expf(m[r] - mnew);
      const float p0 = __expf(s[0][r] - mnew);
      const float p1 = __expf(s[1][r] - mnew);
      s[0][r] = p0; s[1][r] = p1;
      float ps = p0 + p1;
#pragma unroll
      for (int off = 1; off < 16; off <<= 1)
        ps += __shfl_xor(ps, off, 32);
      l[r] = l[r] * alpha[r] + ps;
      m[r] = mnew;
    }
#pragma unroll
    for (int t = 0; t < 4; ++t)
#pragma unroll
      for (int r = 0; r < 8; ++r)
        accO[t][r] *= alpha[r];
#pragma unroll
    for (int r = 0; r < 8; ++r) {
      const int prow = hi * 8 + r;
      P[prow * 32 + l15]      = f2bf(s[0][r]);
      P[prow * 32 + 16 + l15] = f2bf(s[1][r]);
    }
    __syncthreads();
    FragB pa;                               // A-layout 16x32
    pa.q[0] = *(const uint4*)&P[l15 * 32 + koffA];
    pa.q[1] = *(const uint4*)&P[l15 * 32 + koffA + 16];
#pragma unroll
    for (int nt = 0; nt < 4; ++nt) {
      FragB vb;                             // B-layout 32x16 from xvT
      const u16* p = xvT + vbase + (size_t)(nt * 16 + l15) * S_ + j + koffB;
      vb.q[0] = *(const uint4*)(p);
      vb.q[1] = *(const uint4*)(p + 16);
      accO[nt] = __builtin_amdgcn_wmma_f32_16x16x32_bf16(
          false, pa.v, false, vb.v, (short)0, accO[nt], false, false);
    }
    __syncthreads();
  }

#pragma unroll
  for (int r = 0; r < 8; ++r) l[r] = 1.0f / l[r];
#pragma unroll
  for (int nt = 0; nt < 4; ++nt) {
#pragma unroll
    for (int r = 0; r < 8; ++r) {
      const size_t row = (size_t)b * S_ + qt * 16 + hi * 8 + r;
      const int col = colH + nt * 16 + l15;
      ao[row * E_ + col] = f2bf(accO[nt][r] * l[r]);
    }
  }
}

// ---------------------------------------------------------------------------
// Host orchestration
// ---------------------------------------------------------------------------
extern "C" void kernel_launch(void* const* d_in, const int* in_sizes, int n_in,
                              void* d_out, int out_size, void* d_ws, size_t ws_size,
                              hipStream_t stream) {
  (void)in_sizes; (void)n_in; (void)out_size; (void)ws_size;
  const float* inputs  = (const float*)d_in[0];
  const float* proj_w  = (const float*)d_in[1];
  const float* attn_w  = (const float*)d_in[2];
  const float* value_w = (const float*)d_in[3];
  const float* out_w   = (const float*)d_in[4];
  const float* out_b   = (const float*)d_in[5];
  const float* ln1_s   = (const float*)d_in[6];
  const float* ln1_b   = (const float*)d_in[7];
  const float* ln2_s   = (const float*)d_in[8];
  const float* ln2_b   = (const float*)d_in[9];
  const float* w1      = (const float*)d_in[10];
  const float* b1      = (const float*)d_in[11];
  const float* w2      = (const float*)d_in[12];
  const float* b2      = (const float*)d_in[13];
  float* out = (float*)d_out;

  char* wp = (char*)d_ws;
  auto carve = [&](size_t bytes) -> void* {
    void* p = (void*)wp;
    wp += (bytes + 255) & ~(size_t)255;
    return p;
  };
  u16* xbf    = (u16*)carve((size_t)BS_ * E_ * 2);   // LN1 output
  u16* xpbf   = (u16*)carve((size_t)BS_ * E_ * 2);   // projections (== keys)
  u16* xqbf   = (u16*)carve((size_t)BS_ * E_ * 2);   // queries
  u16* xvbf   = (u16*)carve((size_t)BS_ * E_ * 2);   // values [B,S,H,D]
  u16* xvT    = (u16*)carve((size_t)BS_ * E_ * 2);   // values [B,H,D,S]
  u16* aobf   = (u16*)carve((size_t)BS_ * E_ * 2);   // attention output
  u16* ybf    = (u16*)carve((size_t)BS_ * E_ * 2);   // LN2 output
  u16* h1bf   = (u16*)carve((size_t)BS_ * F_ * 2);   // GELU(y@w1+b1)
  float* xres = (float*)carve((size_t)BS_ * E_ * 4); // residual stream
  // transposed bf16 weights [N][K]
  u16* pwT  = (u16*)carve((size_t)E_ * E_ * 2);
  u16* awT  = (u16*)carve((size_t)H_ * D_ * D_ * 2);
  u16* vwT  = (u16*)carve((size_t)H_ * D_ * D_ * 2);
  u16* owT  = (u16*)carve((size_t)E_ * E_ * 2);
  u16* w1T  = (u16*)carve((size_t)E_ * F_ * 2);
  u16* w2T  = (u16*)carve((size_t)F_ * E_ * 2);

  auto cvtT = [&](const float* src, u16* dst, int K, int N, int total) {
    cvt_bf16_t<<<(total + 255) / 256, 256, 0, stream>>>(src, dst, K, N, total);
  };
  cvtT(proj_w,  pwT, E_, E_, E_ * E_);
  cvtT(attn_w,  awT, D_, D_, H_ * D_ * D_);
  cvtT(value_w, vwT, D_, D_, H_ * D_ * D_);
  cvtT(out_w,   owT, E_, E_, E_ * E_);
  cvtT(w1,      w1T, E_, F_, E_ * F_);
  cvtT(w2,      w2T, F_, E_, F_ * E_);

  // LN1
  ln_bf16<<<BS_, 256, 0, stream>>>(inputs, ln1_s, ln1_b, xbf);

  // xp = x @ proj_w   [8192,768] x [768,768] -> bf16
  gemm_bf16<false, false, false, true><<<dim3(E_ / 64, BS_ / 128, 1), 256, 0, stream>>>(
      xbf, E_, 0, pwT, E_, 0, nullptr, nullptr, 0, 0,
      xpbf, nullptr, E_, 0, BS_, E_, E_);

  // xq[h] = xp[:,h] @ attn_w[h]   batched 12x [8192,64]x[64,64]
  gemm_bf16<false, false, false, true><<<dim3(1, BS_ / 128, H_), 256, 0, stream>>>(
      xpbf, E_, D_, awT, D_, (long long)D_ * D_, nullptr, nullptr, 0, 0,
      xqbf, nullptr, E_, D_, BS_, D_, D_);

  // xv[h] = xp[:,h] @ value_w[h]
  gemm_bf16<false, false, false, true><<<dim3(1, BS_ / 128, H_), 256, 0, stream>>>(
      xpbf, E_, D_, vwT, D_, (long long)D_ * D_, nullptr, nullptr, 0, 0,
      xvbf, nullptr, E_, D_, BS_, D_, D_);

  // xv -> xvT [B,H,D,S]
  xv_transpose<<<(BS_ * E_) / 256, 256, 0, stream>>>(xvbf, xvT);

  // flash attention -> aobf
  attn_flash<<<dim3(S_ / 16, H_, B_), 32, 0, stream>>>(xqbf, xpbf, xvT, aobf);

  // x_res = ao @ out_w + out_b + inputs   (f32 out)
  gemm_bf16<true, false, true, false><<<dim3(E_ / 64, BS_ / 128, 1), 256, 0, stream>>>(
      aobf, E_, 0, owT, E_, 0, out_b, inputs, E_, 0,
      nullptr, xres, E_, 0, BS_, E_, E_);

  // LN2
  ln_bf16<<<BS_, 256, 0, stream>>>(xres, ln2_s, ln2_b, ybf);

  // h1 = gelu(y @ w1 + b1)   [8192,768]x[768,3072] -> bf16
  gemm_bf16<true, true, false, true><<<dim3(F_ / 64, BS_ / 128, 1), 256, 0, stream>>>(
      ybf, E_, 0, w1T, E_, 0, b1, nullptr, 0, 0,
      h1bf, nullptr, F_, 0, BS_, F_, E_);

  // out = h1 @ w2 + b2 + x_res   [8192,3072]x[3072,768] -> f32
  gemm_bf16<true, false, true, false><<<dim3(E_ / 64, BS_ / 128, 1), 256, 0, stream>>>(
      h1bf, F_, 0, w2T, F_, 0, b2, xres, E_, 0,
      nullptr, out, E_, 0, BS_, E_, F_);
}
